// SparseSelfAttention_46686294507605
// MI455X (gfx1250) — matrix operand order repaired and checked by reading, not compiled, not run on hardware
//
#include <hip/hip_runtime.h>
#include <hip/hip_bf16.h>

#define S_LEN 2048
#define NH    256
#define KQ    1024
#define HEADS 16
#define HD    64

typedef __attribute__((ext_vector_type(16))) _Float16 v16h;
typedef __attribute__((ext_vector_type(8)))  float    v8f;
typedef __attribute__((ext_vector_type(4)))  int      v4i;

union Frag16 { v16h h; uint4 u[2]; };

static __device__ __forceinline__ v8f wmma_f16(const Frag16& a, const Frag16& b, v8f c) {
    return __builtin_amdgcn_wmma_f32_16x16x32_f16(false, a.h, false, b.h, (short)0, c, false, false);
}

// ---- CDNA5 async global->LDS copy path (ASYNCcnt), with synchronous fallback ----
#if defined(__has_builtin)
#  if __has_builtin(__builtin_amdgcn_global_load_async_to_lds_b128) && \
      __has_builtin(__builtin_amdgcn_s_wait_asynccnt)
#    define USE_ASYNC_LDS 1
#  endif
#endif
#ifndef USE_ASYNC_LDS
#  define USE_ASYNC_LDS 0
#endif

static __device__ __forceinline__ void cp128(_Float16* l, const _Float16* g) {
#if USE_ASYNC_LDS
    // builtin signature: (v4i addrspace(1)*, v4i addrspace(3)*, imm offset, imm cpol)
    __attribute__((address_space(1))) void* gv =
        (__attribute__((address_space(1))) void*)(void*)g;
    __attribute__((address_space(3))) void* lv =
        (__attribute__((address_space(3))) void*)(void*)l;
    __builtin_amdgcn_global_load_async_to_lds_b128(
        (__attribute__((address_space(1))) v4i*)gv,
        (__attribute__((address_space(3))) v4i*)lv, 0, 0);
#else
    *(uint4*)l = *(const uint4*)g;
#endif
}
static __device__ __forceinline__ void cp_fence() {
#if USE_ASYNC_LDS
    __builtin_amdgcn_s_wait_asynccnt(0);
#endif
}

// ---------------- elementwise f32 -> f16 convert ----------------
__global__ __launch_bounds__(256) void cvt_kernel(const float* __restrict__ src,
                                                  _Float16* __restrict__ dst, int n) {
    int i = blockIdx.x * 256 + threadIdx.x;
    if (i < n) dst[i] = (_Float16)src[i];
}

// ---------------- f32 [R][C] -> f16 [C][R] transpose-convert ----------------
__global__ __launch_bounds__(256) void cvt_t_kernel(const float* __restrict__ src,
                                                    _Float16* __restrict__ dst, int R, int C) {
    int i = blockIdx.x * 256 + threadIdx.x;
    if (i < R * C) {
        int r = i / C, c = i % C;
        dst[(size_t)c * R + r] = (_Float16)src[i];
    }
}

// ---------------- LayerNorm (one block per token, 256 threads == NH) ----------------
__global__ __launch_bounds__(256) void ln_kernel(const float* __restrict__ x,
                                                 const float* __restrict__ s,
                                                 const float* __restrict__ b,
                                                 _Float16* __restrict__ y) {
    __shared__ float red[256];
    int row = blockIdx.x, t = threadIdx.x;
    float v = x[(size_t)row * NH + t];
    red[t] = v; __syncthreads();
    for (int o = 128; o > 0; o >>= 1) { if (t < o) red[t] += red[t + o]; __syncthreads(); }
    float mean = red[0] * (1.0f / NH);
    __syncthreads();
    float d = v - mean;
    red[t] = d * d; __syncthreads();
    for (int o = 128; o > 0; o >>= 1) { if (t < o) red[t] += red[t + o]; __syncthreads(); }
    float var = red[0] * (1.0f / NH);
    y[(size_t)row * NH + t] = (_Float16)(d * rsqrtf(var + 1e-5f) * s[t] + b[t]);
}

// ---------------- coordinate MLP: 2 -> 32 -> 32 -> 1024 (silu, silu, linear) ----------------
__global__ __launch_bounds__(256) void mlp_kernel(const float* __restrict__ e,
                                                  const float* __restrict__ w1, const float* __restrict__ b1,
                                                  const float* __restrict__ w2, const float* __restrict__ b2,
                                                  const float* __restrict__ w3, const float* __restrict__ b3,
                                                  float* __restrict__ out) {
    __shared__ float h1[32], h2[32];
    int tok = blockIdx.x, t = threadIdx.x;
    float e0 = e[(size_t)tok * 2], e1 = e[(size_t)tok * 2 + 1];
    if (t < 32) {
        float a = e0 * w1[t] + e1 * w1[32 + t] + b1[t];
        h1[t] = a / (1.0f + __expf(-a));
    }
    __syncthreads();
    if (t < 32) {
        float a = b2[t];
        for (int i = 0; i < 32; ++i) a += h1[i] * w2[i * 32 + t];
        h2[t] = a / (1.0f + __expf(-a));
    }
    __syncthreads();
    for (int j = t; j < KQ; j += 256) {
        float a = b3[j];
        for (int i = 0; i < 32; ++i) a += h2[i] * w3[i * KQ + j];
        out[(size_t)tok * KQ + j] = a;
    }
}

// ---------------- WMMA GEMM: C[M,N] = A[M,K](f16) @ Bt[N,K](f16)^T + bias (+extra) (*alpha)
// block tile 64x64, 128 threads (4 waves), wave tile 16x64, double-buffered async LDS staging
__global__ __launch_bounds__(128) void gemm_kernel(const _Float16* __restrict__ A,
                                                   const _Float16* __restrict__ Bt, // [N][K]
                                                   const float* __restrict__ bias,
                                                   const float* __restrict__ extra,
                                                   const float* __restrict__ alpha,
                                                   _Float16* __restrict__ outH,
                                                   float* __restrict__ outF,
                                                   int M, int N, int K, int transOut) {
    __shared__ _Float16 As[2][64 * 48];   // [row][k], stride 48 halves (96B, 16B multiple)
    __shared__ _Float16 Bs[2][64 * 48];   // [n][k]
    int tid = threadIdx.x, lane = tid & 31, wv = tid >> 5;
    int lm = lane & 15, hi = lane >> 4;
    int kb8 = hi * 8, kb16 = hi * 16;
    int n0 = blockIdx.x * 64, m0 = blockIdx.y * 64;

    v8f acc0 = {}, acc1 = {}, acc2 = {}, acc3 = {};

    auto stage = [&](int b, int kc) {
#pragma unroll
        for (int ii = 0; ii < 2; ++ii) {                 // A tile: 256 x b128
            int i = tid + ii * 128;
            int r = i >> 2, c = (i & 3) * 8;
            cp128(&As[b][r * 48 + c], &A[(size_t)(m0 + r) * K + kc + c]);
        }
#pragma unroll
        for (int ii = 0; ii < 2; ++ii) {                 // B tile: 256 x b128 (rows of Bt)
            int i = tid + ii * 128;
            int r = i >> 2, c = (i & 3) * 8;
            cp128(&Bs[b][r * 48 + c], &Bt[(size_t)(n0 + r) * K + kc + c]);
        }
    };

    int buf = 0;
    stage(0, 0);
    for (int kc = 0; kc < K; kc += 32) {
        cp_fence();
        __syncthreads();                  // tile `buf` ready; prior compute on buf^1 done
        if (kc + 32 < K) stage(buf ^ 1, kc + 32);

        Frag16 a;
        int rb = (wv * 16 + lm) * 48;
        a.u[0] = *(uint4*)&As[buf][rb + kb16];
        a.u[1] = *(uint4*)&As[buf][rb + kb16 + 8];
        Frag16 b;
        b.u[0] = *(uint4*)&Bs[buf][(0 * 16 + lm) * 48 + kb16];
        b.u[1] = *(uint4*)&Bs[buf][(0 * 16 + lm) * 48 + kb16 + 8];
        acc0 = wmma_f16(a, b, acc0);
        b.u[0] = *(uint4*)&Bs[buf][(1 * 16 + lm) * 48 + kb16];
        b.u[1] = *(uint4*)&Bs[buf][(1 * 16 + lm) * 48 + kb16 + 8];
        acc1 = wmma_f16(a, b, acc1);
        b.u[0] = *(uint4*)&Bs[buf][(2 * 16 + lm) * 48 + kb16];
        b.u[1] = *(uint4*)&Bs[buf][(2 * 16 + lm) * 48 + kb16 + 8];
        acc2 = wmma_f16(a, b, acc2);
        b.u[0] = *(uint4*)&Bs[buf][(3 * 16 + lm) * 48 + kb16];
        b.u[1] = *(uint4*)&Bs[buf][(3 * 16 + lm) * 48 + kb16 + 8];
        acc3 = wmma_f16(a, b, acc3);
        buf ^= 1;
    }

    float am = alpha ? alpha[0] : 1.0f;
    v8f accs[4] = {acc0, acc1, acc2, acc3};
    for (int nt = 0; nt < 4; ++nt) {
        for (int v = 0; v < 8; ++v) {
            int mg = m0 + wv * 16 + v + 8 * hi;
            int ng = n0 + nt * 16 + lm;
            float val = accs[nt][v] + bias[ng];
            if (extra) val += extra[(size_t)mg * N + ng];
            val *= am;
            size_t idx = transOut ? ((size_t)ng * M + mg) : ((size_t)mg * N + ng);
            if (outH) outH[idx] = (_Float16)val;
            else      outF[idx] = val;
        }
    }
}

// ---------------- flash attention: grid (S/64, H), 128 threads (4 waves x 16 query rows)
// V pre-transposed to [KQ][S] so every LDS staging is a straight b128 row copy.
__global__ __launch_bounds__(128) void attn_kernel(const _Float16* __restrict__ Q,
                                                   const _Float16* __restrict__ Km,
                                                   const _Float16* __restrict__ Vt,  // [KQ][S]
                                                   const float* __restrict__ bias,
                                                   _Float16* __restrict__ O) {
    __shared__ _Float16 Qs[64 * 72];         // [q][d]
    __shared__ _Float16 Ks[2][32 * 72];      // [key][d]  (QK^T B-frag pairs along d)
    __shared__ _Float16 Vs[2][64 * 48];      // [d][key]  (PV  B-frag pairs along key)
    __shared__ _Float16 Ps[4][16 * 32];      // per-wave prob tile, row-major [q][key]
    int tid = threadIdx.x, lane = tid & 31, wv = tid >> 5;
    int lm = lane & 15, hi = lane >> 4;
    int kb8 = hi * 8, kb16 = hi * 16;
    int q0 = blockIdx.x * 64, h = blockIdx.y;
    const float scale = 0.125f;              // D^-0.5, D=64

#pragma unroll
    for (int ii = 0; ii < 4; ++ii) {         // Q tile once: 512 x b128
        int i = tid + ii * 128;
        int r = i >> 3, c = (i & 7) * 8;
        cp128(&Qs[r * 72 + c], &Q[(size_t)(q0 + r) * KQ + h * HD + c]);
    }
    auto stageKV = [&](int b, int kc) {
#pragma unroll
        for (int ii = 0; ii < 2; ++ii) {     // K tile: 32 rows x 64 halves
            int i = tid + ii * 128;
            int r = i >> 3, c = (i & 7) * 8;
            cp128(&Ks[b][r * 72 + c], &Km[(size_t)(kc + r) * KQ + h * HD + c]);
        }
#pragma unroll
        for (int ii = 0; ii < 2; ++ii) {     // Vt tile: 64 rows(d) x 32 halves(key)
            int i = tid + ii * 128;
            int d = i >> 2, c = (i & 3) * 8;
            cp128(&Vs[b][d * 48 + c], &Vt[(size_t)(h * HD + d) * S_LEN + kc + c]);
        }
    };
    stageKV(0, 0);

    Frag16 aq0, aq1;
    v8f o0 = {}, o1 = {}, o2 = {}, o3 = {};
    float rmax[8], rsum[8];
    for (int v = 0; v < 8; ++v) { rmax[v] = -1e30f; rsum[v] = 0.0f; }
    int qrow = q0 + wv * 16 + 8 * hi;

    int buf = 0;
    for (int kc = 0; kc < S_LEN; kc += 32) {
        cp_fence();
        __syncthreads();                      // tile `buf` (and Qs on first iter) ready
        if (kc == 0) {
            int rb = (wv * 16 + lm) * 72;
            aq0.u[0] = *(uint4*)&Qs[rb + kb16];      aq0.u[1] = *(uint4*)&Qs[rb + kb16 + 8];
            aq1.u[0] = *(uint4*)&Qs[rb + 32 + kb16]; aq1.u[1] = *(uint4*)&Qs[rb + 32 + kb16 + 8];
        }
        if (kc + 32 < S_LEN) stageKV(buf ^ 1, kc + 32);

        // scores: two 16x16 key tiles, K-dim d=64 -> 2 WMMA each
        v8f s0 = {}, s1 = {};
        {
            Frag16 b;
            b.u[0] = *(uint4*)&Ks[buf][lm * 72 + kb16];            b.u[1] = *(uint4*)&Ks[buf][lm * 72 + kb16 + 8];
            s0 = wmma_f16(aq0, b, s0);
            b.u[0] = *(uint4*)&Ks[buf][lm * 72 + 32 + kb16];       b.u[1] = *(uint4*)&Ks[buf][lm * 72 + 32 + kb16 + 8];
            s0 = wmma_f16(aq1, b, s0);
            b.u[0] = *(uint4*)&Ks[buf][(16 + lm) * 72 + kb16];     b.u[1] = *(uint4*)&Ks[buf][(16 + lm) * 72 + kb16 + 8];
            s1 = wmma_f16(aq0, b, s1);
            b.u[0] = *(uint4*)&Ks[buf][(16 + lm) * 72 + 32 + kb16];b.u[1] = *(uint4*)&Ks[buf][(16 + lm) * 72 + 32 + kb16 + 8];
            s1 = wmma_f16(aq1, b, s1);
        }

        // streamed bias + online softmax (C-layout rows live in 16-lane halves)
        const float* bh = bias + ((size_t)h * S_LEN + (size_t)qrow) * S_LEN + kc;
        __builtin_prefetch(&bh[32], 0, 1);    // next chunk of the 268MB bias stream
        for (int v = 0; v < 8; ++v) {
            float x0 = s0[v] * scale + bh[(size_t)v * S_LEN + lm];
            float x1 = s1[v] * scale + bh[(size_t)v * S_LEN + 16 + lm];
            float mloc = fmaxf(x0, x1);
            for (int off = 8; off > 0; off >>= 1) mloc = fmaxf(mloc, __shfl_xor(mloc, off, 16));
            float nm = fmaxf(rmax[v], mloc);
            float corr = __expf(rmax[v] - nm);
            float p0 = __expf(x0 - nm), p1 = __expf(x1 - nm);
            float ls = p0 + p1;
            for (int off = 8; off > 0; off >>= 1) ls += __shfl_xor(ls, off, 16);
            rsum[v] = rsum[v] * corr + ls;
            rmax[v] = nm;
            o0[v] *= corr; o1[v] *= corr; o2[v] *= corr; o3[v] *= corr;
            s0[v] = p0; s1[v] = p1;
        }

        // re-layout P: C-frag -> wave-private LDS (row-major) -> A-frag (same-wave DS ordering)
        _Float16* pp = &Ps[wv][0];
        for (int v = 0; v < 8; ++v) {
            pp[(v + 8 * hi) * 32 + lm]      = (_Float16)s0[v];
            pp[(v + 8 * hi) * 32 + 16 + lm] = (_Float16)s1[v];
        }
        Frag16 pa;
        pa.u[0] = *(uint4*)&pp[lm * 32 + kb16];
        pa.u[1] = *(uint4*)&pp[lm * 32 + kb16 + 8];

        Frag16 bv;
        bv.u[0] = *(uint4*)&Vs[buf][(0 * 16 + lm) * 48 + kb16]; bv.u[1] = *(uint4*)&Vs[buf][(0 * 16 + lm) * 48 + kb16 + 8];
        o0 = wmma_f16(pa, bv, o0);
        bv.u[0] = *(uint4*)&Vs[buf][(1 * 16 + lm) * 48 + kb16]; bv.u[1] = *(uint4*)&Vs[buf][(1 * 16 + lm) * 48 + kb16 + 8];
        o1 = wmma_f16(pa, bv, o1);
        bv.u[0] = *(uint4*)&Vs[buf][(2 * 16 + lm) * 48 + kb16]; bv.u[1] = *(uint4*)&Vs[buf][(2 * 16 + lm) * 48 + kb16 + 8];
        o2 = wmma_f16(pa, bv, o2);
        bv.u[0] = *(uint4*)&Vs[buf][(3 * 16 + lm) * 48 + kb16]; bv.u[1] = *(uint4*)&Vs[buf][(3 * 16 + lm) * 48 + kb16 + 8];
        o3 = wmma_f16(pa, bv, o3);
        buf ^= 1;
    }

    for (int v = 0; v < 8; ++v) {
        int mg = qrow + v;
        float inv = 1.0f / rsum[v];
        O[(size_t)mg * KQ + h * HD + 0 * 16 + lm] = (_Float16)(o0[v] * inv);
        O[(size_t)mg * KQ + h * HD + 1 * 16 + lm] = (_Float16)(o1[v] * inv);
        O[(size_t)mg * KQ + h * HD + 2 * 16 + lm] = (_Float16)(o2[v] * inv);
        O[(size_t)mg * KQ + h * HD + 3 * 16 + lm] = (_Float16)(o3[v] * inv);
    }
}

extern "C" void kernel_launch(void* const* d_in, const int* in_sizes, int n_in,
                              void* d_out, int out_size, void* d_ws, size_t ws_size,
                              hipStream_t stream) {
    const float* x        = (const float*)d_in[0];
    const float* emb      = (const float*)d_in[1];
    const float* attnbias = (const float*)d_in[2];
    const float* ln_s[3]  = {(const float*)d_in[3],  (const float*)d_in[7],  (const float*)d_in[11]};
    const float* ln_b[3]  = {(const float*)d_in[4],  (const float*)d_in[8],  (const float*)d_in[12]};
    const float* w[3]     = {(const float*)d_in[5],  (const float*)d_in[9],  (const float*)d_in[13]};
    const float* bvec[3]  = {(const float*)d_in[6],  (const float*)d_in[10], (const float*)d_in[14]};
    const float* mw1[3]   = {(const float*)d_in[15], (const float*)d_in[21], (const float*)d_in[27]};
    const float* mb1[3]   = {(const float*)d_in[16], (const float*)d_in[22], (const float*)d_in[28]};
    const float* mw2[3]   = {(const float*)d_in[17], (const float*)d_in[23], (const float*)d_in[29]};
    const float* mb2[3]   = {(const float*)d_in[18], (const float*)d_in[24], (const float*)d_in[30]};
    const float* mw3[3]   = {(const float*)d_in[19], (const float*)d_in[25], (const float*)d_in[31]};
    const float* mb3[3]   = {(const float*)d_in[20], (const float*)d_in[26], (const float*)d_in[32]};
    const float* out_w    = (const float*)d_in[33];
    const float* out_b    = (const float*)d_in[34];
    const float* alpha    = (const float*)d_in[35];

    size_t off = 0;
    auto carve = [&](size_t bytes) {
        void* p = (char*)d_ws + off;
        off += (bytes + 255) & ~(size_t)255;
        return p;
    };
    _Float16* xln[3];  for (int i = 0; i < 3; ++i) xln[i]  = (_Float16*)carve((size_t)S_LEN * NH * 2);
    _Float16* wfT[3];  for (int i = 0; i < 3; ++i) wfT[i]  = (_Float16*)carve((size_t)NH * KQ * 2);   // [KQ][NH]
    _Float16* owfT    = (_Float16*)carve((size_t)KQ * NH * 2);                                        // [NH][KQ]
    float*    mlp[3];  for (int i = 0; i < 3; ++i) mlp[i]  = (float*)carve((size_t)S_LEN * KQ * 4);
    _Float16* qkv[3];  for (int i = 0; i < 3; ++i) qkv[i]  = (_Float16*)carve((size_t)S_LEN * KQ * 2); // v is [KQ][S]
    _Float16* of16    = (_Float16*)carve((size_t)S_LEN * KQ * 2);

    // 1) weight convert + transpose to f16 [N][K]
    for (int i = 0; i < 3; ++i)
        cvt_t_kernel<<<(NH * KQ + 255) / 256, 256, 0, stream>>>(w[i], wfT[i], NH, KQ);
    cvt_t_kernel<<<(KQ * NH + 255) / 256, 256, 0, stream>>>(out_w, owfT, KQ, NH);

    // 2) layernorms + coordinate MLPs
    for (int i = 0; i < 3; ++i) {
        ln_kernel<<<S_LEN, 256, 0, stream>>>(x, ln_s[i], ln_b[i], xln[i]);
        mlp_kernel<<<S_LEN, 256, 0, stream>>>(emb, mw1[i], mb1[i], mw2[i], mb2[i], mw3[i], mb3[i], mlp[i]);
    }

    // 3) q/k/v = ln(x) @ W + b + mlp  (WMMA, f16 out; v written transposed [KQ][S])
    for (int i = 0; i < 3; ++i)
        gemm_kernel<<<dim3(KQ / 64, S_LEN / 64), 128, 0, stream>>>(
            xln[i], wfT[i], bvec[i], mlp[i], nullptr, qkv[i], nullptr,
            S_LEN, KQ, NH, /*transOut=*/(i == 2) ? 1 : 0);

    // 4) flash attention with streamed bias
    attn_kernel<<<dim3(S_LEN / 64, HEADS), 128, 0, stream>>>(
        qkv[0], qkv[1], qkv[2], attnbias, of16);

    // 5) output projection: (o @ out_w + out_b) * alpha -> f32 d_out
    gemm_kernel<<<dim3(NH / 64, S_LEN / 64), 128, 0, stream>>>(
        of16, owfT, out_b, nullptr, alpha, nullptr, (float*)d_out,
        S_LEN, NH, KQ, /*transOut=*/0);
}
